// Goat_67559835566259
// MI455X (gfx1250) — matrix operand order repaired and checked by reading, not compiled, not run on hardware
//
#include <hip/hip_runtime.h>
#include <hip/hip_bf16.h>

typedef __attribute__((ext_vector_type(16))) _Float16 v16h;
typedef __attribute__((ext_vector_type(8)))  _Float16 v8h;
typedef __attribute__((ext_vector_type(8)))  float    v8f;

#define NBV    50     // neighborhood size
#define DIMV   128    // embedding dim
#define MP     64     // padded neighborhood (4 WMMA tiles of 16)
#define F16STR 136    // halves per LDS row (128 + 8 pad -> conflict-free b128 loads)
#define ALSTR  68     // floats per align row (64 + 4 pad -> 16B-aligned rows, conflict-free)

// Branch-free tanh: prefer the gfx1250 hardware trans-op.
__device__ __forceinline__ float fast_tanhf(float x) {
#if __has_builtin(__builtin_amdgcn_tanhf)
    return __builtin_amdgcn_tanhf(x);            // v_tanh_f32
#else
    // tanh(x) = sign(x) * (1 - e) / (1 + e),  e = exp(-2|x|)   (branch-free)
    const float e = __expf(-2.0f * __builtin_fabsf(x));   // v_exp_f32 path
    const float t = (1.0f - e) / (1.0f + e);
    return __builtin_copysignf(t, x);
#endif
}

__launch_bounds__(256)
__global__ void goat_attn_kernel(const int*   __restrict__ snbr,
                                 const int*   __restrict__ tnbr,
                                 const float* __restrict__ smask,
                                 const float* __restrict__ tmask,
                                 const float* __restrict__ emb,
                                 float*       __restrict__ out,
                                 int B)
{
    __shared__ _Float16 sf16[MP * F16STR];   // src tile, f16, padded rows zeroed
    __shared__ _Float16 tf16[MP * F16STR];   // tgt tile
    __shared__ float    align_s[MP * ALSTR]; // tanh(S @ T^T), 64x64 (+4 pad)
    __shared__ int      sidx[NBV];
    __shared__ int      tidx[NBV];
    __shared__ float    rowm[MP], colm[MP], se[MP], te[MP], srcw[MP], trgw[MP];

    const int b   = blockIdx.x;
    const int tid = threadIdx.x;

    // ---- stage neighbor indices ----
    if (tid < NBV)                           sidx[tid]      = snbr[(size_t)b * NBV + tid];
    else if (tid >= 64 && tid < 64 + NBV)    tidx[tid - 64] = tnbr[(size_t)b * NBV + (tid - 64)];

    // ---- zero pad rows 50..63 (cols 0..127) so WMMA tail tiles are clean ----
    for (int i = tid; i < 14 * 64; i += 256) {
        int r  = NBV + (i >> 6);
        int c2 = (i & 63) << 1;              // half index, step 2
        *(uint32_t*)&sf16[r * F16STR + c2] = 0u;
        *(uint32_t*)&tf16[r * F16STR + c2] = 0u;
    }
    __syncthreads();

    // ---- gather: 50 rows x 32 float4 chunks per matrix, convert fp32 -> f16 ----
    for (int i = tid; i < NBV * 32; i += 256) {
        const int r  = i >> 5;
        const int c4 = i & 31;
        {
            const float4 v = *(const float4*)(emb + (size_t)sidx[r] * DIMV + c4 * 4);
            union { _Float16 h[4]; uint2 u; } pk;
            pk.h[0] = (_Float16)v.x; pk.h[1] = (_Float16)v.y;
            pk.h[2] = (_Float16)v.z; pk.h[3] = (_Float16)v.w;
            *(uint2*)&sf16[r * F16STR + c4 * 4] = pk.u;
        }
        {
            const float4 v = *(const float4*)(emb + (size_t)tidx[r] * DIMV + c4 * 4);
            union { _Float16 h[4]; uint2 u; } pk;
            pk.h[0] = (_Float16)v.x; pk.h[1] = (_Float16)v.y;
            pk.h[2] = (_Float16)v.z; pk.h[3] = (_Float16)v.w;
            *(uint2*)&tf16[r * F16STR + c4 * 4] = pk.u;
        }
    }
    __syncthreads();

    // ---- WMMA: align = S (64x128) @ T^T (128x64), 16 tiles, 8 waves x 2 tiles ----
    const int wave = tid >> 5;
    const int lane = tid & 31;
    const int lhi  = lane >> 4;   // lane half: 0 or 1
    const int llo  = lane & 15;

    #pragma unroll
    for (int t = 0; t < 2; ++t) {
        const int tile = wave * 2 + t;
        const int mi = tile >> 2, ni = tile & 3;
        v8f acc = {};
        // A fragment base: row = mi*16 + llo; K interleave {0..7,16..23} / {8..15,24..31}
        const _Float16* aBase = &sf16[(mi * 16 + llo) * F16STR + (lhi << 3)];
        // B fragment: column n = ni*16 + llo of T^T == row n of T; K interleave {0..7,16..23} / {8..15,24..31}
        const _Float16* bBase = &tf16[(ni * 16 + llo) * F16STR];
        #pragma unroll
        for (int ks = 0; ks < 4; ++ks) {
            v16h a, bm;
            *(v8h*)&a       = *(const v8h*)(aBase + ks * 32);        // K k..k+7 (or +8 shifted)
            *((v8h*)&a + 1) = *(const v8h*)(aBase + ks * 32 + 16);   // K k+16..k+23
            const int koffB = ks * 32 + lhi * 8;                     // K k..k+7 and k+16..k+23 (same as A)
            *(v8h*)&bm       = *(const v8h*)(bBase + koffB);
            *((v8h*)&bm + 1) = *(const v8h*)(bBase + koffB + 16);
            acc = __builtin_amdgcn_wmma_f32_16x16x32_f16(
                      false, a, false, bm, (short)0, acc, false, false);
        }
        // C/D layout: VGPR i -> M = mi*16 + i + lhi*8 ; N = ni*16 + llo
        #pragma unroll
        for (int i = 0; i < 8; ++i) {
            const int row = mi * 16 + i + lhi * 8;
            const int col = ni * 16 + llo;
            align_s[row * ALSTR + col] = fast_tanhf(acc[i]);
        }
    }
    __syncthreads();

    // ---- means over opposite neighborhood (+ masks) ----
    if (tid < NBV) {
        // row mean: vectorized float4 reads (row base is 16B aligned, stride 272B)
        const float4* rp = (const float4*)&align_s[tid * ALSTR];
        float s = 0.f;
        #pragma unroll
        for (int q = 0; q < 12; ++q) {       // cols 0..47
            const float4 v = rp[q];
            s += v.x + v.y + v.z + v.w;
        }
        s += align_s[tid * ALSTR + 48] + align_s[tid * ALSTR + 49];
        rowm[tid] = s * (1.0f / NBV) + smask[(size_t)b * NBV + tid];
    } else if (tid >= 64 && tid < 64 + NBV) {
        const int t = tid - 64;
        float s = 0.f;
        for (int ss = 0; ss < NBV; ++ss) s += align_s[ss * ALSTR + t];
        colm[t] = s * (1.0f / NBV) + tmask[(size_t)b * NBV + t];
    }
    __syncthreads();

    // ---- softmax pass 1: exp(x - max) ----
    if (tid < NBV) {
        float mx = -1e30f;
        for (int j = 0; j < NBV; ++j) mx = fmaxf(mx, rowm[j]);   // broadcast LDS reads
        se[tid] = __expf(rowm[tid] - mx);
    } else if (tid >= 64 && tid < 64 + NBV) {
        const int t = tid - 64;
        float mx = -1e30f;
        for (int j = 0; j < NBV; ++j) mx = fmaxf(mx, colm[j]);
        te[t] = __expf(colm[t] - mx);
    }
    __syncthreads();

    // ---- softmax pass 2: normalize ----
    if (tid < NBV) {
        float dsum = 0.f;
        for (int j = 0; j < NBV; ++j) dsum += se[j];
        srcw[tid] = se[tid] / dsum;
    } else if (tid >= 64 && tid < 64 + NBV) {
        const int t = tid - 64;
        float dsum = 0.f;
        for (int j = 0; j < NBV; ++j) dsum += te[j];
        trgw[t] = te[t] / dsum;
    }
    __syncthreads();

    // ---- weighted context sums -> [B,128] each, coalesced store ----
    if (tid < 128) {
        const int d = tid;
        float acc = 0.f;
        #pragma unroll 5
        for (int s = 0; s < NBV; ++s) acc += srcw[s] * (float)sf16[s * F16STR + d];
        out[(size_t)b * DIMV + d] = acc;
    } else {
        const int d = tid - 128;
        float acc = 0.f;
        #pragma unroll 5
        for (int s = 0; s < NBV; ++s) acc += trgw[s] * (float)tf16[s * F16STR + d];
        out[(size_t)B * DIMV + (size_t)b * DIMV + d] = acc;
    }
}

extern "C" void kernel_launch(void* const* d_in, const int* in_sizes, int n_in,
                              void* d_out, int out_size, void* d_ws, size_t ws_size,
                              hipStream_t stream) {
    const int*   snbr  = (const int*)d_in[0];
    const int*   tnbr  = (const int*)d_in[1];
    const float* smask = (const float*)d_in[2];
    const float* tmask = (const float*)d_in[3];
    const float* emb   = (const float*)d_in[4];
    float*       out   = (float*)d_out;
    const int B = in_sizes[0] / NBV;   // 16384
    goat_attn_kernel<<<B, 256, 0, stream>>>(snbr, tnbr, smask, tmask, emb, out, B);
}
